// Transformer_3427383903094
// MI455X (gfx1250) — compile-verified, matrix operand order checked by reading, not run on hardware
//
#include <hip/hip_runtime.h>

#define DIMX 1024
#define HEADS 16
#define DHEAD 64
#define LAYERS 4
#define FFDIM 4096
#define BATCH 2
#define SEQ 2048
#define TOKENS (BATCH * SEQ)

typedef __attribute__((ext_vector_type(16))) _Float16 v16h;
typedef __attribute__((ext_vector_type(8)))  _Float16 v8h;
typedef __attribute__((ext_vector_type(8)))  float    v8f;
typedef int v4i __attribute__((__vector_size__(16)));  // matches builtin proto

// Load one 16-bit A/B-style WMMA fragment: caller points p at
// [row][kbase] (kbase = 8*(lane>=16)); halves 0..7 = K kbase..kbase+7,
// halves 8..15 = K kbase+16..kbase+23 (ISA 7.12.2 16-bit A layout).
__device__ __forceinline__ v16h frag_ld(const _Float16* p) {
  v8h lo = *(const v8h*)(p);
  v8h hi = *(const v8h*)(p + 16);
  v16h r;
#pragma unroll
  for (int i = 0; i < 8; ++i) { r[i] = lo[i]; r[i + 8] = hi[i]; }
  return r;
}

__device__ __forceinline__ v8f wmma_f16(v16h a, v16h b, v8f c) {
  return __builtin_amdgcn_wmma_f32_16x16x32_f16(false, a, false, b,
                                                (short)0, c, false, false);
}

// ---- CDNA5 async global->LDS copy (ASYNCcnt path), 16 bytes per lane ------
__device__ __forceinline__ void async_ld_b128(const void* gsrc, void* ldst) {
#if __has_builtin(__builtin_amdgcn_global_load_async_to_lds_b128)
  typedef __attribute__((address_space(1))) v4i gv4i;
  typedef __attribute__((address_space(3))) v4i lv4i;
  __builtin_amdgcn_global_load_async_to_lds_b128(
      (gv4i*)(unsigned long long)(size_t)gsrc,
      (lv4i*)(unsigned)(size_t)ldst, 0, 0);
#else
  unsigned loff = (unsigned)(size_t)ldst;          // LDS byte offset (addr[31:0])
  unsigned long long ga = (unsigned long long)(size_t)gsrc;
  asm volatile("global_load_async_to_lds_b128 %0, %1, off"
               :: "v"(loff), "v"(ga) : "memory");
#endif
}

__device__ __forceinline__ void wait_async0() {
#if __has_builtin(__builtin_amdgcn_s_wait_asynccnt)
  __builtin_amdgcn_s_wait_asynccnt(0);
#else
  asm volatile("s_wait_asynccnt 0" ::: "memory");
#endif
}

// ---------------- weight fp32 -> f16 transpose: Wt[n*K+k] = W[k*N+n] --------
__global__ __launch_bounds__(256) void wtrans_k(const float* __restrict__ W,
                                                _Float16* __restrict__ Wt,
                                                int K, int N) {
  size_t idx = (size_t)blockIdx.x * 256 + threadIdx.x;
  int n = (int)(idx / K);
  int k = (int)(idx % K);
  Wt[idx] = (_Float16)W[(size_t)k * N + n];
}

// ---------------- RMSNorm: one row (DIMX) per block ------------------------
__global__ __launch_bounds__(256) void rmsnorm_k(const float* __restrict__ x,
                                                 const float* __restrict__ g,
                                                 _Float16* __restrict__ out16,
                                                 float* __restrict__ out32) {
  const int row = blockIdx.x;
  const float* xr = x + (size_t)row * DIMX;
  float s = 0.0f;
#pragma unroll
  for (int i = threadIdx.x; i < DIMX; i += 256) { float v = xr[i]; s += v * v; }
#pragma unroll
  for (int m = 16; m >= 1; m >>= 1) s += __shfl_xor(s, m, 32);
  __shared__ float red[8];
  if ((threadIdx.x & 31) == 0) red[threadIdx.x >> 5] = s;
  __syncthreads();
  float tot = 0.0f;
#pragma unroll
  for (int i = 0; i < 8; ++i) tot += red[i];
  const float norm = sqrtf(tot) * 0.03125f;  // * DIMX^-0.5 (1024^-0.5)
  const float inv = 1.0f / fmaxf(norm, 1e-8f);
#pragma unroll
  for (int i = threadIdx.x; i < DIMX; i += 256) {
    float y = xr[i] * inv * g[i];
    if (out16) out16[(size_t)row * DIMX + i] = (_Float16)y;
    if (out32) out32[(size_t)row * DIMX + i] = y;
  }
}

// ---------------- WMMA GEMM: C[M,N] = A[M,K] * Bt[N,K]^T -------------------
// 128x128 block tile, 8 waves (2x4), wave tile 64x32, K-step 32.
// Double-buffered LDS fed by GLOBAL_LOAD_ASYNC_TO_LDS_B128 (ASYNCcnt).
#define GLD 40  // LDS row stride in halves (32 + 8 pad) -> 80B rows, 16B aligned
template <bool OUT16, bool ACCUM, bool BIAS, bool GELU>
__global__ __launch_bounds__(256) void gemm_f16_k(
    const _Float16* __restrict__ A, const _Float16* __restrict__ Bt,
    const float* __restrict__ bias, _Float16* __restrict__ out16,
    float* __restrict__ accum, int M, int N, int K, float scale) {
  __shared__ _Float16 lAbuf[2][128 * GLD];
  __shared__ _Float16 lBbuf[2][128 * GLD];
  const int tid = threadIdx.x;
  const int lane = tid & 31, wave = tid >> 5;
  const int l16 = lane & 15, lhi = lane >> 4;
  const int wm = wave >> 2, wn = wave & 3;
  const int bm = blockIdx.x * 128;
  const int bn = blockIdx.y * 128;
  const int ldr = tid >> 1;          // 0..127: tile row loaded by this thread
  const int ldk = (tid & 1) * 16;    // 0/16: half-offset within K-step

  v8f acc[4][2] = {};

  auto issue_tile = [&](int kt, int buf) {
    const int k0 = kt * 32;
    const _Float16* ga = A + (size_t)(bm + ldr) * K + k0 + ldk;
    _Float16* la = &lAbuf[buf][ldr * GLD + ldk];
    async_ld_b128(ga, la);
    async_ld_b128(ga + 8, la + 8);
    const _Float16* gb = Bt + (size_t)(bn + ldr) * K + k0 + ldk;
    _Float16* lb = &lBbuf[buf][ldr * GLD + ldk];
    async_ld_b128(gb, lb);
    async_ld_b128(gb + 8, lb + 8);
  };

  const int nT = K >> 5;
  issue_tile(0, 0);
  for (int t = 0; t < nT; ++t) {
    const int cur = t & 1;
    wait_async0();          // this wave's tile-t DMA complete
    __syncthreads();        // every wave's tile-t DMA complete
    if (t + 1 < nT) issue_tile(t + 1, cur ^ 1);  // overlaps with compute below

    v16h af[4], bf[2];
#pragma unroll
    for (int ms = 0; ms < 4; ++ms)
      af[ms] = frag_ld(&lAbuf[cur][(wm * 64 + ms * 16 + l16) * GLD + lhi * 8]);
#pragma unroll
    for (int ns = 0; ns < 2; ++ns)
      bf[ns] = frag_ld(&lBbuf[cur][(wn * 32 + ns * 16 + l16) * GLD + lhi * 8]);
#pragma unroll
    for (int ms = 0; ms < 4; ++ms)
#pragma unroll
      for (int ns = 0; ns < 2; ++ns)
        acc[ms][ns] = wmma_f16(af[ms], bf[ns], acc[ms][ns]);
    __syncthreads();        // done reading buf[cur] before it is overwritten
  }

#pragma unroll
  for (int ms = 0; ms < 4; ++ms) {
#pragma unroll
    for (int ns = 0; ns < 2; ++ns) {
      const int col = bn + wn * 32 + ns * 16 + l16;
      float bv = 0.0f;
      if constexpr (BIAS) bv = bias[col];
#pragma unroll
      for (int r = 0; r < 8; ++r) {
        const int row = bm + wm * 64 + ms * 16 + lhi * 8 + r;
        float e = acc[ms][ns][r] * scale + bv;
        if constexpr (GELU) e = 0.5f * e * (1.0f + erff(e * 0.70710678118f));
        const size_t idx = (size_t)row * N + col;
        if constexpr (OUT16) out16[idx] = (_Float16)e;
        if constexpr (ACCUM) accum[idx] += e;
      }
    }
  }
}

// ---------------- Flash attention (causal, ALiBi, MQA: 1 KV head) ----------
// grid (SEQ/128, HEADS, BATCH), 256 threads; wave w owns q rows [i0+16w,+16).
#define LDK 72  // K tile row stride in halves (64+8) -> 144B
#define LDV 40  // Vt row stride (32+8) -> 80B
#define LDP 40
__global__ __launch_bounds__(256) void attn_k(const _Float16* __restrict__ q16,
                                              const _Float16* __restrict__ kv16,
                                              _Float16* __restrict__ out16) {
  const int i0 = blockIdx.x * 128;
  const int h = blockIdx.y;
  const int b = blockIdx.z;
  const int tid = threadIdx.x, lane = tid & 31, wave = tid >> 5;
  const int l16 = lane & 15, lhi = lane >> 4;
  __shared__ _Float16 lK[32 * LDK];
  __shared__ _Float16 lVt[64 * LDV];
  __shared__ _Float16 lP[8][16 * LDP];

  const float slope = exp2f(-0.5f * (float)(h + 1));
  const int irow0 = i0 + wave * 16;

  v16h qf[2];
  {
    const int row = irow0 + l16;
    const _Float16* qp = q16 + (size_t)(b * SEQ + row) * (HEADS * DHEAD) + h * DHEAD;
    qf[0] = frag_ld(qp + lhi * 8);        // d chunk 0..31
    qf[1] = frag_ld(qp + 32 + lhi * 8);   // d chunk 32..63
  }

  v8f o[4] = {};
  float mrow[8], lrow[8];
#pragma unroll
  for (int r = 0; r < 8; ++r) { mrow[r] = -1e30f; lrow[r] = 0.0f; }

  for (int j0 = 0; j0 < i0 + 128; j0 += 32) {
    // cooperative: K tile (32 x 64) row-major, V tile transposed (64 x 32)
    const int jj = tid >> 3, doff = (tid & 7) * 8;
    const _Float16* kvrow = kv16 + (size_t)(b * SEQ + j0 + jj) * (2 * DHEAD);
    *(v8h*)&lK[jj * LDK + doff] = *(const v8h*)(kvrow + doff);
    v8h vv = *(const v8h*)(kvrow + DHEAD + doff);
#pragma unroll
    for (int u = 0; u < 8; ++u) lVt[(doff + u) * LDV + jj] = vv[u];
    __syncthreads();

    v8f s[2] = {};
#pragma unroll
    for (int nt = 0; nt < 2; ++nt) {
      v16h kf0 = frag_ld(&lK[(nt * 16 + l16) * LDK + lhi * 8]);
      v16h kf1 = frag_ld(&lK[(nt * 16 + l16) * LDK + 32 + lhi * 8]);
      s[nt] = wmma_f16(qf[0], kf0, s[nt]);
      s[nt] = wmma_f16(qf[1], kf1, s[nt]);
    }

#pragma unroll
    for (int r = 0; r < 8; ++r) {
      const int irow = irow0 + lhi * 8 + r;
      float v0 = s[0][r] + slope * (float)(j0 + l16);
      float v1 = s[1][r] + slope * (float)(j0 + 16 + l16);
      if (j0 + l16 > irow)      v0 = -1e30f;
      if (j0 + 16 + l16 > irow) v1 = -1e30f;
      float vmax = fmaxf(v0, v1);
#pragma unroll
      for (int m = 8; m >= 1; m >>= 1) vmax = fmaxf(vmax, __shfl_xor(vmax, m, 32));
      const float mn = fmaxf(mrow[r], vmax);
      const float al = __expf(mrow[r] - mn);
      const float p0 = __expf(v0 - mn);
      const float p1 = __expf(v1 - mn);
      float rs = p0 + p1;
#pragma unroll
      for (int m = 8; m >= 1; m >>= 1) rs += __shfl_xor(rs, m, 32);
      lrow[r] = lrow[r] * al + rs;
      mrow[r] = mn;
#pragma unroll
      for (int t = 0; t < 4; ++t) o[t][r] *= al;
      lP[wave][(lhi * 8 + r) * LDP + l16]      = (_Float16)p0;
      lP[wave][(lhi * 8 + r) * LDP + 16 + l16] = (_Float16)p1;
    }

    // P (16x32) in A-layout from per-wave LDS (in-order DS ops within a wave)
    v16h pf = frag_ld(&lP[wave][l16 * LDP + lhi * 8]);
#pragma unroll
    for (int t = 0; t < 4; ++t) {
      v16h vf = frag_ld(&lVt[(t * 16 + l16) * LDV + lhi * 8]);
      o[t] = wmma_f16(pf, vf, o[t]);
    }
    __syncthreads();
  }

#pragma unroll
  for (int t = 0; t < 4; ++t) {
#pragma unroll
    for (int r = 0; r < 8; ++r) {
      const int irow = irow0 + lhi * 8 + r;
      const float e = o[t][r] / lrow[r];
      out16[(size_t)(b * SEQ + irow) * (HEADS * DHEAD) + h * DHEAD + t * 16 + l16] =
          (_Float16)e;
    }
  }
}

// ---------------------------------------------------------------------------
extern "C" void kernel_launch(void* const* d_in, const int* in_sizes, int n_in,
                              void* d_out, int out_size, void* d_ws, size_t ws_size,
                              hipStream_t stream) {
  (void)in_sizes; (void)n_in; (void)out_size; (void)ws_size;
  const float* x       = (const float*)d_in[0];
  const float* Wq      = (const float*)d_in[1];   // [L,1024,1024]
  const float* Wkv     = (const float*)d_in[2];   // [L,1024,128]
  const float* Wo      = (const float*)d_in[3];   // [L,1024,1024]
  const float* g_attn  = (const float*)d_in[4];   // [L,1024]
  const float* Wf1     = (const float*)d_in[5];   // [L,1024,4096]
  const float* bf1     = (const float*)d_in[6];   // [L,4096]
  const float* Wf2     = (const float*)d_in[7];   // [L,4096,1024]
  const float* bf2     = (const float*)d_in[8];   // [L,1024]
  const float* g_ff    = (const float*)d_in[9];   // [L,1024]
  const float* g_final = (const float*)d_in[10];  // [1024]
  float* out = (float*)d_out;

  char* p = (char*)d_ws;
  auto alloc = [&](size_t bytes) {
    char* r = p; p += (bytes + 255) & ~(size_t)255; return r;
  };
  float*     xbuf   = (float*)alloc((size_t)TOKENS * DIMX * 4);
  _Float16*  h16    = (_Float16*)alloc((size_t)TOKENS * DIMX * 2);
  _Float16*  q16    = (_Float16*)alloc((size_t)TOKENS * DIMX * 2);
  _Float16*  kv16   = (_Float16*)alloc((size_t)TOKENS * 2 * DHEAD * 2);
  _Float16*  attn16 = (_Float16*)alloc((size_t)TOKENS * DIMX * 2);
  _Float16*  g16    = (_Float16*)alloc((size_t)TOKENS * FFDIM * 2);
  _Float16*  wq16   = (_Float16*)alloc((size_t)DIMX * DIMX * 2);
  _Float16*  wkv16  = (_Float16*)alloc((size_t)DIMX * 2 * DHEAD * 2);
  _Float16*  wo16   = (_Float16*)alloc((size_t)DIMX * DIMX * 2);
  _Float16*  wf116  = (_Float16*)alloc((size_t)DIMX * FFDIM * 2);
  _Float16*  wf216  = (_Float16*)alloc((size_t)FFDIM * DIMX * 2);

  (void)hipMemcpyAsync(xbuf, x, (size_t)TOKENS * DIMX * 4,
                       hipMemcpyDeviceToDevice, stream);

  for (int l = 0; l < LAYERS; ++l) {
    // weight convert+transpose into f16 Wt[N][K]
    wtrans_k<<<(DIMX * DIMX) / 256, 256, 0, stream>>>(
        Wq + (size_t)l * DIMX * DIMX, wq16, DIMX, DIMX);
    wtrans_k<<<(DIMX * 2 * DHEAD) / 256, 256, 0, stream>>>(
        Wkv + (size_t)l * DIMX * 2 * DHEAD, wkv16, DIMX, 2 * DHEAD);
    wtrans_k<<<(DIMX * DIMX) / 256, 256, 0, stream>>>(
        Wo + (size_t)l * DIMX * DIMX, wo16, DIMX, DIMX);
    wtrans_k<<<(DIMX * FFDIM) / 256, 256, 0, stream>>>(
        Wf1 + (size_t)l * DIMX * FFDIM, wf116, DIMX, FFDIM);
    wtrans_k<<<(FFDIM * DIMX) / 256, 256, 0, stream>>>(
        Wf2 + (size_t)l * FFDIM * DIMX, wf216, FFDIM, DIMX);

    // attention branch
    rmsnorm_k<<<TOKENS, 256, 0, stream>>>(xbuf, g_attn + (size_t)l * DIMX, h16, nullptr);
    gemm_f16_k<true, false, false, false>
        <<<dim3(TOKENS / 128, DIMX / 128), 256, 0, stream>>>(
        h16, wq16, nullptr, q16, nullptr, TOKENS, DIMX, DIMX, 0.125f);  // q * d^-1/2
    gemm_f16_k<true, false, false, false>
        <<<dim3(TOKENS / 128, 1), 256, 0, stream>>>(
        h16, wkv16, nullptr, kv16, nullptr, TOKENS, 2 * DHEAD, DIMX, 1.0f);
    attn_k<<<dim3(SEQ / 128, HEADS, BATCH), 256, 0, stream>>>(q16, kv16, attn16);
    gemm_f16_k<false, true, false, false>
        <<<dim3(TOKENS / 128, DIMX / 128), 256, 0, stream>>>(
        attn16, wo16, nullptr, nullptr, xbuf, TOKENS, DIMX, DIMX, 1.0f);  // x += out@Wo

    // FF branch
    rmsnorm_k<<<TOKENS, 256, 0, stream>>>(xbuf, g_ff + (size_t)l * DIMX, h16, nullptr);
    gemm_f16_k<true, false, true, true>
        <<<dim3(TOKENS / 128, FFDIM / 128), 256, 0, stream>>>(
        h16, wf116, bf1 + (size_t)l * FFDIM, g16, nullptr, TOKENS, FFDIM, DIMX, 1.0f);
    gemm_f16_k<false, true, true, false>
        <<<dim3(TOKENS / 128, DIMX / 128), 256, 0, stream>>>(
        g16, wf216, bf2 + (size_t)l * DIMX, nullptr, xbuf, TOKENS, DIMX, FFDIM, 1.0f);
  }

  rmsnorm_k<<<TOKENS, 256, 0, stream>>>(xbuf, g_final, nullptr, out);
}
